// PatchCoherentLoss_33629593927680
// MI455X (gfx1250) — compile-verified, matrix operand order checked by reading, not run on hardware
//
#include <hip/hip_runtime.h>

typedef __attribute__((ext_vector_type(16))) _Float16 v16h;
typedef __attribute__((ext_vector_type(8)))  _Float16 v8h;
typedef __attribute__((ext_vector_type(8)))  float    v8f;

// Problem constants: x,y are [2,3,112,112] fp32; 7x7 patches stride 1.
#define OW      106            // 112-7+1
#define NPATCH  (OW*OW)        // 11236 patches per image
#define DFEAT   147            // 3*7*7
#define KPAD    160            // K padded to multiple of 32 (5 WMMA k-steps)
#define NPAD    11264          // 44 * 256  (columns, x patches)
#define MPAD    11248          // 703 * 16  (rows, y patches)
#define MITER   (MPAD/16)      // 703
#define NTILES  44             // blocks in N per image (256 cols each)
#define NPARTS  (NTILES*2)
#define LDSROW  168            // LDS row stride in halves (336 B, conflict-free)

// ---------------------------------------------------------------------------
// Kernel 1: img [2,3,112,112] fp32 -> padded fp16 patch matrix [2,nPad,KPAD]
//           + fp32 squared norms [2,nPad]. Pad rows: zeros, norm = padNorm.
// ---------------------------------------------------------------------------
__global__ void build_patches(const float* __restrict__ img,
                              _Float16* __restrict__ out,
                              float* __restrict__ norms,
                              int nReal, int nPad, float padNorm)
{
    int tid = blockIdx.x * blockDim.x + threadIdx.x;
    if (tid >= 2 * nPad) return;
    int b = tid / nPad;
    int r = tid - b * nPad;

    _Float16* orow = out + ((size_t)b * nPad + r) * KPAD;
    if (r >= nReal) {
        for (int k = 0; k < KPAD; ++k) orow[k] = (_Float16)0.0f;
        norms[(size_t)b * nPad + r] = padNorm;
        return;
    }
    int oy = r / OW, ox = r - oy * OW;
    const float* ib = img + (size_t)b * 3 * 112 * 112;
    float nrm = 0.0f;
    int k = 0;
    for (int c = 0; c < 3; ++c)
        for (int ky = 0; ky < 7; ++ky)
            for (int kx = 0; kx < 7; ++kx) {
                float v = ib[(c * 112 + oy + ky) * 112 + ox + kx];
                nrm += v * v;
                orow[k++] = (_Float16)v;
            }
    for (; k < KPAD; ++k) orow[k] = (_Float16)0.0f;
    norms[(size_t)b * nPad + r] = nrm;
}

// ---------------------------------------------------------------------------
// Async-prefetch one 16-row yp tile (16 x 320 B) into LDS (row stride 336 B).
// Issued by one wave (32 lanes x 16 B x 10 instr = 5120 B), ASYNCcnt-tracked.
// ---------------------------------------------------------------------------
__device__ __forceinline__ void prefetch_tile_async(unsigned ldsbase,
                                                    const _Float16* __restrict__ gsrc,
                                                    int lane)
{
#pragma unroll
    for (int i = 0; i < 10; ++i) {
        int ch  = i * 32 + lane;          // 0..319 : 16-byte chunk index
        int row = ch / 20;                // 20 chunks per 320-byte row
        int c16 = ch - row * 20;
        unsigned lds = ldsbase + (unsigned)(row * (LDSROW * 2) + c16 * 16);
        const _Float16* g = gsrc + (size_t)row * KPAD + c16 * 8;
        asm volatile("global_load_async_to_lds_b128 %0, %1, off"
                     :: "v"(lds), "v"(g) : "memory");
    }
}

// ---------------------------------------------------------------------------
// SSD between x-patch n and y-patch m, read from the original fp32 images.
// ---------------------------------------------------------------------------
__device__ __forceinline__ float patch_ssd(const float* __restrict__ xb,
                                           const float* __restrict__ yb,
                                           int n, int m)
{
    if (n >= NPATCH) return 0.0f;
    int oyx = n / OW, oxx = n - oyx * OW;
    int oym = m / OW, oxm = m - oym * OW;
    float ssd = 0.0f;
    for (int c = 0; c < 3; ++c)
        for (int ky = 0; ky < 7; ++ky)
#pragma unroll
            for (int kx = 0; kx < 7; ++kx) {
                float xv = xb[(c * 112 + oyx + ky) * 112 + oxx + kx];
                float yv = yb[(c * 112 + oym + ky) * 112 + oxm + kx];
                float df = xv - yv;
                ssd += df * df;
            }
    return ssd;
}

// ---------------------------------------------------------------------------
// Kernel 2: fused dist-GEMM (WMMA f16->f32) + fp16 argmin over rows + SSD.
// Block = 256 threads = 8 waves; wave w owns 32 columns (2 x 16-col WMMA
// tiles -> two independent WMMA chains). The shared 16-row yp tile is
// double-buffered in LDS via async global->LDS copies issued by wave 0.
// ---------------------------------------------------------------------------
__global__ __launch_bounds__(256, 1)
void dist_argmin_ssd(const _Float16* __restrict__ xp,
                     const _Float16* __restrict__ ypm,
                     const float* __restrict__ xn,
                     const float* __restrict__ yn,
                     const float* __restrict__ xi,
                     const float* __restrict__ yi,
                     float* __restrict__ partials)
{
    __shared__ _Float16 atile[2][16 * LDSROW];   // 2 x 5.25 KB double buffer
    __shared__ float    sred[256];

    const int lane  = threadIdx.x & 31;
    const int wv    = threadIdx.x >> 5;
    const int b     = blockIdx.y;
    const int nbase = blockIdx.x * 256;
    const int ncol0 = nbase + wv * 32 + (lane & 15);
    const int ncol1 = ncol0 + 16;
    // 16-bit A/B fragment layout: lanes 0-15 hold K {0..7,16..23}+k0,
    // lanes 16-31 hold K {8..15,24..31}+k0 of their row (A) / column (B).
    const int koff   = (lane < 16) ? 0 : 8;
    const int rowoff = (lane < 16) ? 0 : 8;

    // B fragments (x patches, this wave's 2x16 columns) resident all loop.
    v16h bfrag0[5], bfrag1[5];
    {
        const _Float16* b0 = xp + ((size_t)b * NPAD + ncol0) * KPAD + koff;
        const _Float16* b1 = xp + ((size_t)b * NPAD + ncol1) * KPAD + koff;
#pragma unroll
        for (int s = 0; s < 5; ++s) {
            union { v16h v; v8h h[2]; } u0, u1;
            u0.h[0] = *(const v8h*)(b0 + s * 32);
            u0.h[1] = *(const v8h*)(b0 + s * 32 + 16);
            u1.h[0] = *(const v8h*)(b1 + s * 32);
            u1.h[1] = *(const v8h*)(b1 + s * 32 + 16);
            bfrag0[s] = u0.v;
            bfrag1[s] = u1.v;
        }
    }
    const float xnorm0 = xn[(size_t)b * NPAD + ncol0];
    const float xnorm1 = xn[(size_t)b * NPAD + ncol1];
    const float* ynb = yn + (size_t)b * MPAD;
    const _Float16* ypb = ypm + (size_t)b * MPAD * KPAD;

    const unsigned ldsbuf0 = (unsigned)(uintptr_t)&atile[0][0];
    const unsigned ldsbuf1 = (unsigned)(uintptr_t)&atile[1][0];

    // Prologue: stage tile 0 into LDS buffer 0.
    if (wv == 0) {
        prefetch_tile_async(ldsbuf0, ypb, lane);
        asm volatile("s_wait_asynccnt 0x0" ::: "memory");
    }
    __syncthreads();

    float bestV0 = __builtin_huge_valf(), bestV1 = __builtin_huge_valf();
    int   bestI0 = 0, bestI1 = 0;

    for (int it = 0; it < MITER; ++it) {
        const int cur = it & 1;
        const int m0  = it * 16;

        // Wave 0 prefetches the next tile into the other buffer.
        if (wv == 0 && it + 1 < MITER)
            prefetch_tile_async(cur ? ldsbuf0 : ldsbuf1,
                                ypb + (size_t)(m0 + 16) * KPAD, lane);

        // A fragments from LDS (ds_load_b128, conflict-free padded rows).
        const _Float16* ap = &atile[cur][(lane & 15) * LDSROW + koff];
        v16h afrag[5];
#pragma unroll
        for (int s = 0; s < 5; ++s) {
            union { v16h v; v8h h[2]; } u;
            u.h[0] = *(const v8h*)(ap + s * 32);
            u.h[1] = *(const v8h*)(ap + s * 32 + 16);
            afrag[s] = u.v;
        }

        // Two independent WMMA accumulation chains (hides XDL latency).
        v8f acc0 = {}, acc1 = {};
#pragma unroll
        for (int s = 0; s < 5; ++s) {
            acc0 = __builtin_amdgcn_wmma_f32_16x16x32_f16(
                false, afrag[s], false, bfrag0[s], (short)0, acc0, false, false);
            acc1 = __builtin_amdgcn_wmma_f32_16x16x32_f16(
                false, afrag[s], false, bfrag1[s], (short)0, acc1, false, false);
        }

        // Running fp16 argmin per column (row index increases monotonically).
#pragma unroll
        for (int r = 0; r < 8; ++r) {
            int m = m0 + rowoff + r;
            float ynm = ynb[m];
            float d0 = (float)(_Float16)(ynm + xnorm0 - 2.0f * acc0[r]);
            float d1 = (float)(_Float16)(ynm + xnorm1 - 2.0f * acc1[r]);
            if (d0 < bestV0) { bestV0 = d0; bestI0 = m; }
            if (d1 < bestV1) { bestV1 = d1; bestI1 = m; }
        }

        if (wv == 0)
            asm volatile("s_wait_asynccnt 0x0" ::: "memory");
        __syncthreads();
    }

    // Merge the two half-wave partners holding the same column
    // (rows split 0..7 / 8..15 per tile). Tie-break: smaller row index.
    {
        float oV0 = __shfl_xor(bestV0, 16, 32);
        int   oI0 = __shfl_xor(bestI0, 16, 32);
        if (oV0 < bestV0 || (oV0 == bestV0 && oI0 < bestI0)) { bestV0 = oV0; bestI0 = oI0; }
        float oV1 = __shfl_xor(bestV1, 16, 32);
        int   oI1 = __shfl_xor(bestI1, 16, 32);
        if (oV1 < bestV1 || (oV1 == bestV1 && oI1 < bestI1)) { bestV1 = oV1; bestI1 = oI1; }
    }

    if (lane < 16) {
        const float* xb = xi + (size_t)b * 3 * 112 * 112;
        const float* yb = yi + (size_t)b * 3 * 112 * 112;
        sred[wv * 32 + (lane & 15)]      = patch_ssd(xb, yb, ncol0, bestI0);
        sred[wv * 32 + 16 + (lane & 15)] = patch_ssd(xb, yb, ncol1, bestI1);
    }
    __syncthreads();
    if (threadIdx.x == 0) {
        float s = 0.0f;
        for (int i = 0; i < 256; ++i) s += sred[i];  // fixed order: deterministic
        partials[blockIdx.y * gridDim.x + blockIdx.x] = s;
    }
}

// ---------------------------------------------------------------------------
// Kernel 3: deterministic final reduction + mean normalization.
// loss = total_ssd / (batch * N * d)
// ---------------------------------------------------------------------------
__global__ void finalize_kernel(const float* __restrict__ partials,
                                float* __restrict__ out)
{
    if (threadIdx.x == 0 && blockIdx.x == 0) {
        float s = 0.0f;
        for (int i = 0; i < NPARTS; ++i) s += partials[i];
        out[0] = s / (2.0f * (float)NPATCH * (float)DFEAT);
    }
}

// ---------------------------------------------------------------------------
extern "C" void kernel_launch(void* const* d_in, const int* in_sizes, int n_in,
                              void* d_out, int out_size, void* d_ws, size_t ws_size,
                              hipStream_t stream)
{
    (void)in_sizes; (void)n_in; (void)out_size; (void)ws_size;
    const float* x = (const float*)d_in[0];
    const float* y = (const float*)d_in[1];

    char* ws = (char*)d_ws;
    size_t off = 0;
    auto take = [&](size_t bytes) -> char* {
        char* p = ws + off;
        off = (off + bytes + 255) & ~(size_t)255;
        return p;
    };
    _Float16* xp16  = (_Float16*)take((size_t)2 * NPAD * KPAD * sizeof(_Float16));
    _Float16* yp16  = (_Float16*)take((size_t)2 * MPAD * KPAD * sizeof(_Float16));
    float*    xnorm = (float*)   take((size_t)2 * NPAD * sizeof(float));
    float*    ynorm = (float*)   take((size_t)2 * MPAD * sizeof(float));
    float*    parts = (float*)   take((size_t)NPARTS * sizeof(float));

    build_patches<<<(2 * NPAD + 255) / 256, 256, 0, stream>>>(
        x, xp16, xnorm, NPATCH, NPAD, 0.0f);
    build_patches<<<(2 * MPAD + 255) / 256, 256, 0, stream>>>(
        y, yp16, ynorm, NPATCH, MPAD, 1e30f);   // pad rows never win argmin

    dim3 grid(NTILES, 2);
    dist_argmin_ssd<<<grid, 256, 0, stream>>>(
        xp16, yp16, xnorm, ynorm, x, y, parts);

    finalize_kernel<<<1, 1, 0, stream>>>(parts, (float*)d_out);
}
